// MAGClassifier_21277267984744
// MI455X (gfx1250) — matrix operand order repaired and checked by reading, not compiled, not run on hardware
//
#include <hip/hip_runtime.h>
#include <stdint.h>

// ---------------------------------------------------------------------------
// CDNA5 (gfx1250) wave32 WMMA types & helpers
// ---------------------------------------------------------------------------
typedef __attribute__((ext_vector_type(16))) __bf16 v16bf;
typedef __attribute__((ext_vector_type(8)))  float  v8f;

__device__ __forceinline__ v8f zerov8() {
    v8f z = {0.f, 0.f, 0.f, 0.f, 0.f, 0.f, 0.f, 0.f};
    return z;
}

__device__ __forceinline__ v8f wmma_bf16(v16bf a, v16bf b, v8f c) {
    // D(f32,16x16) = A(bf16,16x32) * B(bf16,32x16) + C
    return __builtin_amdgcn_wmma_f32_16x16x32_bf16(
        /*neg_a=*/false, a, /*neg_b=*/false, b,
        /*c_mod=*/(short)0, c, /*reuse_a=*/false, /*reuse_b=*/false);
}

// Async global->LDS copy of 16 bytes per lane (ASYNCcnt-tracked, no VGPR data).
__device__ __forceinline__ void async_b128(void* lds, const void* g) {
    uint32_t l = (uint32_t)(uintptr_t)lds;  // low 32 bits = LDS byte offset
    asm volatile("global_load_async_to_lds_b128 %0, %1, off"
                 :: "v"(l), "v"(g)
                 : "memory");
}
__device__ __forceinline__ void async_wait0() {
    asm volatile("s_wait_asynccnt 0x0" ::: "memory");
}

// A-fragment (16x32 bf16, MxK) from LDS, row-major with row stride ldk.
// lane(m = lane&15, half = lane>>4) holds K = {half*8+0..7, 16+half*8+0..7}.
__device__ __forceinline__ v16bf load_A(const __bf16* a, int ldk) {
    int lane = threadIdx.x & 31;
    int m  = lane & 15;
    int hf = lane >> 4;
    const __bf16* row = a + m * ldk;
    v16bf r;
#pragma unroll
    for (int e = 0; e < 16; ++e) {
        int k = ((e & 8) << 1) + hf * 8 + (e & 7);
        r[e] = row[k];
    }
    return r;
}

// B-fragment (32x16 bf16, KxN) from LDS where the tile is stored TRANSPOSED
// (N rows of K elements, row stride ldk). lane(n=lane&15) holds K = {kg*8+0..7, 16+kg*8+0..7}.
__device__ __forceinline__ v16bf load_B(const __bf16* bt, int ldk) {
    int lane = threadIdx.x & 31;
    int nn = lane & 15;
    int kg = lane >> 4;
    const __bf16* row = bt + nn * ldk + kg * 8;
    v16bf r;
#pragma unroll
    for (int e = 0; e < 16; ++e) r[e] = row[((e & 8) << 1) + (e & 7)];
    return r;
}

// ---------------------------------------------------------------------------
// Problem constants
// ---------------------------------------------------------------------------
#define NB   32
#define EPG  512
#define STOT (NB * EPG)  // 16384
#define HD   128
#define NHD  8
#define DH   16

// ---------------------------------------------------------------------------
// Kernel 0: weight convert + transpose: in[K x 128] f32 -> out[128 x Kp] bf16
// (zero-padded K..Kp). Makes all GEMM weight staging a raw async copy.
// ---------------------------------------------------------------------------
__global__ __launch_bounds__(256)
void k_wcvt(const float* __restrict__ in, __bf16* __restrict__ out, int K, int Kp) {
    int i = blockIdx.x * 256 + threadIdx.x;
    if (i >= 128 * Kp) return;
    int n = i / Kp, k = i % Kp;
    out[i] = (__bf16)((k < K) ? in[k * 128 + n] : 0.f);
}

// ---------------------------------------------------------------------------
// Kernel 1: gather edge features + fused 2-layer input MLP (WMMA).
// Weights pre-transposed bf16 (w1t[128][160], w2t[128][128]), staged async.
// block: 256 threads (8 waves), 64 rows per block; output bf16.
// ---------------------------------------------------------------------------
__global__ __launch_bounds__(256)
void k_edge_mlp(const float* __restrict__ x, const float* __restrict__ ea,
                const int* __restrict__ src, const int* __restrict__ dst,
                const __bf16* __restrict__ w1t, const float* __restrict__ b1,
                const __bf16* __restrict__ w2t, const float* __restrict__ b2,
                __bf16* __restrict__ out) {
    __shared__ __align__(16) __bf16 As[64][32];
    __shared__ __align__(16) __bf16 Wt[128][32];
    __shared__ __align__(16) __bf16 Hmid[64][128];
    __shared__ float  bias1[128], bias2[128];
    __shared__ __align__(16) int se[64], de[64];

    int tid = threadIdx.x, w = tid >> 5, lane = tid & 31;
    int row0 = blockIdx.x * 64;
    if (tid < 64) { se[tid] = src[row0 + tid]; de[tid] = dst[row0 + tid]; }
    if (tid < 128) { bias1[tid] = b1[tid]; bias2[tid] = b2[tid]; }
    __syncthreads();

    int rt = w & 3, cpar = w >> 2;
    v8f acc[4];
#pragma unroll
    for (int j = 0; j < 4; ++j) acc[j] = zerov8();

    // ---- layer 1: K = 160 (144 real, 16 zero pad) ----
    for (int kk = 0; kk < 160; kk += 32) {
        // A: gather + f32->bf16 convert (can't async this one)
        for (int i = tid; i < 64 * 32; i += 256) {
            int r = i >> 5, c = i & 31, k = kk + c;
            float f;
            if (k < 64)        f = x[se[r] * 64 + k];
            else if (k < 128)  f = x[de[r] * 64 + (k - 64)];
            else if (k < 144)  f = ea[(row0 + r) * 16 + (k - 128)];
            else               f = 0.f;
            As[r][c] = (__bf16)f;
        }
        // W: raw async copy from pre-transposed bf16 arena
        for (int i = tid; i < 128 * 4; i += 256) {
            int n2 = i >> 2, q = i & 3;
            async_b128(&Wt[n2][q * 8], w1t + (size_t)n2 * 160 + kk + q * 8);
        }
        async_wait0();
        __syncthreads();
        v16bf a = load_A(&As[rt * 16][0], 32);
#pragma unroll
        for (int j = 0; j < 4; ++j) {
            int ct = 2 * j + cpar;
            v16bf bb = load_B(&Wt[ct * 16][0], 32);
            acc[j] = wmma_bf16(a, bb, acc[j]);
        }
        __syncthreads();
    }
    int nn = lane & 15, hf = lane >> 4;
#pragma unroll
    for (int j = 0; j < 4; ++j) {
        int ct = 2 * j + cpar;
#pragma unroll
        for (int r = 0; r < 8; ++r) {
            int m = rt * 16 + r + 8 * hf;
            int col = ct * 16 + nn;
            Hmid[m][col] = (__bf16)fmaxf(acc[j][r] + bias1[col], 0.f);
        }
    }
    __syncthreads();

    // ---- layer 2: K = 128 ----
    v8f acc2[4];
#pragma unroll
    for (int j = 0; j < 4; ++j) acc2[j] = zerov8();
    for (int kk = 0; kk < 128; kk += 32) {
        for (int i = tid; i < 128 * 4; i += 256) {
            int n2 = i >> 2, q = i & 3;
            async_b128(&Wt[n2][q * 8], w2t + (size_t)n2 * 128 + kk + q * 8);
        }
        async_wait0();
        __syncthreads();
        v16bf a = load_A(&Hmid[rt * 16][kk], 128);
#pragma unroll
        for (int j = 0; j < 4; ++j) {
            int ct = 2 * j + cpar;
            v16bf bb = load_B(&Wt[ct * 16][0], 32);
            acc2[j] = wmma_bf16(a, bb, acc2[j]);
        }
        __syncthreads();
    }
#pragma unroll
    for (int j = 0; j < 4; ++j) {
        int ct = 2 * j + cpar;
#pragma unroll
        for (int r = 0; r < 8; ++r) {
            int row = row0 + rt * 16 + r + 8 * hf;
            int col = ct * 16 + nn;
            out[(size_t)row * 128 + col] = (__bf16)(acc2[j][r] + bias2[col]);
        }
    }
}

// ---------------------------------------------------------------------------
// Kernel 2: generic WMMA GEMM, bf16 in/out:
//   out[M x 128] = act(A[M x 128] @ W[128 x 128] + bias (+res))
// A and W (pre-transposed bf16) staged with async global->LDS b128 copies.
// block: 256 threads (8 waves), 64 rows per block.
// ---------------------------------------------------------------------------
__global__ __launch_bounds__(256)
void k_gemm(const __bf16* __restrict__ A, const __bf16* __restrict__ wt,
            const float* __restrict__ bias, const __bf16* __restrict__ res,
            __bf16* __restrict__ out, int relu) {
    __shared__ __align__(16) __bf16 As[64][32];
    __shared__ __align__(16) __bf16 Wt[128][32];
    __shared__ float  bs[128];

    int tid = threadIdx.x, w = tid >> 5, lane = tid & 31;
    int row0 = blockIdx.x * 64;
    if (tid < 128) bs[tid] = bias[tid];
    int rt = w & 3, cpar = w >> 2;

    v8f acc[4];
#pragma unroll
    for (int j = 0; j < 4; ++j) acc[j] = zerov8();

    for (int kk = 0; kk < 128; kk += 32) {
        {   // A slab: 64 rows x 64B = 256 x b128, exactly one per thread
            int r = tid >> 2, q = tid & 3;
            async_b128(&As[r][q * 8], A + (size_t)(row0 + r) * 128 + kk + q * 8);
        }
        for (int i = tid; i < 128 * 4; i += 256) {  // W slab: 512 x b128
            int n2 = i >> 2, q = i & 3;
            async_b128(&Wt[n2][q * 8], wt + (size_t)n2 * 128 + kk + q * 8);
        }
        async_wait0();
        __syncthreads();
        v16bf a = load_A(&As[rt * 16][0], 32);
#pragma unroll
        for (int j = 0; j < 4; ++j) {
            int ct = 2 * j + cpar;
            v16bf bb = load_B(&Wt[ct * 16][0], 32);
            acc[j] = wmma_bf16(a, bb, acc[j]);
        }
        __syncthreads();
    }
    int nn = lane & 15, hf = lane >> 4;
#pragma unroll
    for (int j = 0; j < 4; ++j) {
        int ct = 2 * j + cpar;
#pragma unroll
        for (int r = 0; r < 8; ++r) {
            int row = row0 + rt * 16 + r + 8 * hf;
            int col = ct * 16 + nn;
            float v = acc[j][r] + bs[col];
            if (res)  v += (float)res[(size_t)row * 128 + col];
            if (relu) v = fmaxf(v, 0.f);
            out[(size_t)row * 128 + col] = (__bf16)v;
        }
    }
}

// ---------------------------------------------------------------------------
// Kernel 3: flash attention per (graph, head, 64-query tile), WMMA QK^T & P*V.
// Q/K/V are bf16 [S x 128]; K, Q and endpoints staged via async copies.
// Mask recomputed from edge endpoints. block: 128 threads (4 waves).
// ---------------------------------------------------------------------------
__global__ __launch_bounds__(128)
void k_attn(const __bf16* __restrict__ Qb, const __bf16* __restrict__ Kb,
            const __bf16* __restrict__ Vb, const int* __restrict__ src,
            const int* __restrict__ dst, __bf16* __restrict__ Ob, int masked) {
    __shared__ __align__(16) __bf16 Ks[512][32];   // d padded 16->32 with zeros
    __shared__ __align__(16) __bf16 Vt[16][528];   // V^T [d][key], 16 pad cols
    __shared__ __align__(16) int    se[512], de[512];
    __shared__ __align__(16) __bf16 Qs[4][16][32]; // per-wave Q tiles, d pad 32
    __shared__ __align__(16) __bf16 Ps[4][16][32]; // prob scratch, cols16..31=0

    int b = blockIdx.x, h = blockIdx.y, qt = blockIdx.z;
    int tid = threadIdx.x, w = tid >> 5, lane = tid & 31;
    int ebase = b * EPG;
    int col0 = h * DH;
    int qloc0 = qt * 64;

    // async staging: K rows (2x16B each), endpoints, Q rows
    for (int i = tid; i < 1024; i += 128) {
        int key = i >> 1, hl = i & 1;
        async_b128(&Ks[key][hl * 8], Kb + (size_t)(ebase + key) * 128 + col0 + hl * 8);
    }
    {   // 512 ints = 128 chunks each; tid in [0,128)
        async_b128(&se[tid * 4], src + ebase + tid * 4);
        async_b128(&de[tid * 4], dst + ebase + tid * 4);
    }
    for (int i = tid; i < 128; i += 128) {
        int row = i >> 1, hl = i & 1;
        int ww = row >> 4, m = row & 15;
        async_b128(&Qs[ww][m][hl * 8],
                   Qb + (size_t)(ebase + qloc0 + row) * 128 + col0 + hl * 8);
    }
    // VALU staging: V transpose + zero pads
    for (int i = tid; i < 512 * 16; i += 128) {
        int key = i >> 4, d = i & 15;
        Ks[key][16 + d] = (__bf16)0.f;
        Vt[d][key] = Vb[(size_t)(ebase + key) * 128 + col0 + d];
    }
    for (int i = tid; i < 256; i += 128) { int d = i >> 4, c = i & 15; Vt[d][512 + c] = (__bf16)0.f; }
    for (int i = tid; i < 64 * 16; i += 128) {
        int row = i >> 4, c = i & 15;
        int ww = row >> 4, m = row & 15;
        Qs[ww][m][16 + c] = (__bf16)0.f;
        Ps[ww][m][16 + c] = (__bf16)0.f;
    }
    async_wait0();
    __syncthreads();

    v16bf qf = load_A(&Qs[w][0][0], 32);
    v8f o = zerov8();
    float mr[8], lr[8];
#pragma unroll
    for (int r = 0; r < 8; ++r) { mr[r] = -3.0e38f; lr[r] = 0.f; }
    int nn = lane & 15, hf = lane >> 4;
    int qsv[8], qdv[8];
#pragma unroll
    for (int r = 0; r < 8; ++r) {
        int ql = qloc0 + w * 16 + r + 8 * hf;
        qsv[r] = se[ql]; qdv[r] = de[ql];
    }
    const float scale = 0.25f;  // 1/sqrt(16)

    for (int kt = 0; kt < 32; ++kt) {
        v16bf kf = load_B(&Ks[kt * 16][0], 32);
        v8f s = wmma_bf16(qf, kf, zerov8());
        int kkey = kt * 16 + nn;
        int ksv = se[kkey], kdv = de[kkey];
        float pv[8];
#pragma unroll
        for (int r = 0; r < 8; ++r) {
            float sv = s[r] * scale;
            if (masked) {
                bool adj = (qsv[r] == ksv) || (qsv[r] == kdv) ||
                           (qdv[r] == ksv) || (qdv[r] == kdv);
                if (!adj) sv = -1.0e9f;
            }
            float mt = sv;
            mt = fmaxf(mt, __shfl_xor(mt, 1, 32));
            mt = fmaxf(mt, __shfl_xor(mt, 2, 32));
            mt = fmaxf(mt, __shfl_xor(mt, 4, 32));
            mt = fmaxf(mt, __shfl_xor(mt, 8, 32));
            float mn = fmaxf(mr[r], mt);
            float p  = __expf(sv - mn);
            float ps = p;
            ps += __shfl_xor(ps, 1, 32);
            ps += __shfl_xor(ps, 2, 32);
            ps += __shfl_xor(ps, 4, 32);
            ps += __shfl_xor(ps, 8, 32);
            float alpha = __expf(mr[r] - mn);
            lr[r] = lr[r] * alpha + ps;
            o[r] *= alpha;
            mr[r] = mn;
            pv[r] = p;
        }
#pragma unroll
        for (int r = 0; r < 8; ++r) Ps[w][r + 8 * hf][nn] = (__bf16)pv[r];
        v16bf pf = load_A(&Ps[w][0][0], 32);
        v16bf vf = load_B(&Vt[0][kt * 16], 528);
        o = wmma_bf16(pf, vf, o);
    }
#pragma unroll
    for (int r = 0; r < 8; ++r) {
        int qrow = ebase + qloc0 + w * 16 + r + 8 * hf;
        Ob[(size_t)qrow * 128 + col0 + nn] = (__bf16)(o[r] / lr[r]);
    }
}

// ---------------------------------------------------------------------------
// Kernel 4: row-wise LayerNorm (bf16 in/out, f32 stats), one wave per row
// ---------------------------------------------------------------------------
__global__ __launch_bounds__(256)
void k_ln(const __bf16* __restrict__ in, const float* __restrict__ g,
          const float* __restrict__ be, __bf16* __restrict__ out) {
    int w = threadIdx.x >> 5, lane = threadIdx.x & 31;
    int row = blockIdx.x * 8 + w;
    const __bf16* p = in + (size_t)row * 128;
    float v[4];
#pragma unroll
    for (int i = 0; i < 4; ++i) v[i] = (float)p[lane + 32 * i];
    float s = v[0] + v[1] + v[2] + v[3];
#pragma unroll
    for (int off = 1; off < 32; off <<= 1) s += __shfl_xor(s, off, 32);
    float mean = s * (1.f / 128.f);
    float q = 0.f;
#pragma unroll
    for (int i = 0; i < 4; ++i) { float d = v[i] - mean; q += d * d; }
#pragma unroll
    for (int off = 1; off < 32; off <<= 1) q += __shfl_xor(q, off, 32);
    float rs = rsqrtf(q * (1.f / 128.f) + 1e-5f);
    __bf16* o = out + (size_t)row * 128;
#pragma unroll
    for (int i = 0; i < 4; ++i) {
        int c = lane + 32 * i;
        o[c] = (__bf16)((v[i] - mean) * rs * g[c] + be[c]);
    }
}

// ---------------------------------------------------------------------------
// Kernel 5: PMA attention (Sq = 1) per graph — VALU, tiny
// ---------------------------------------------------------------------------
__global__ __launch_bounds__(256)
void k_pma(const float* __restrict__ seed, const float* __restrict__ wq,
           const float* __restrict__ bq, const __bf16* __restrict__ Kp,
           const __bf16* __restrict__ Vp, float* __restrict__ pooled) {
    __shared__ float qp[128];
    __shared__ float sc[8][512];
    int b = blockIdx.x, tid = threadIdx.x;
    if (tid < 128) {
        float a = bq[tid];
        for (int k = 0; k < 128; ++k) a += seed[k] * wq[k * 128 + tid];
        qp[tid] = a;
    }
    __syncthreads();
    for (int i = tid; i < NHD * EPG; i += 256) {
        int hh = i >> 9, key = i & 511;
        const __bf16* kr = Kp + (size_t)(b * EPG + key) * 128 + hh * DH;
        float s = 0.f;
#pragma unroll
        for (int d = 0; d < DH; ++d) s += qp[hh * DH + d] * (float)kr[d];
        sc[hh][key] = s * 0.25f;
    }
    __syncthreads();
    int w = tid >> 5, lane = tid & 31;  // wave w handles head w
    {
        float vals[16];
        float mv = -3.0e38f;
#pragma unroll
        for (int i = 0; i < 16; ++i) { vals[i] = sc[w][lane + 32 * i]; mv = fmaxf(mv, vals[i]); }
#pragma unroll
        for (int off = 1; off < 32; off <<= 1) mv = fmaxf(mv, __shfl_xor(mv, off, 32));
        float ssum = 0.f;
#pragma unroll
        for (int i = 0; i < 16; ++i) { vals[i] = __expf(vals[i] - mv); ssum += vals[i]; }
#pragma unroll
        for (int off = 1; off < 32; off <<= 1) ssum += __shfl_xor(ssum, off, 32);
        float inv = 1.f / ssum;
#pragma unroll
        for (int i = 0; i < 16; ++i) sc[w][lane + 32 * i] = vals[i] * inv;
    }
    __syncthreads();
    if (tid < 128) {
        int hh = tid >> 4, d = tid & 15;
        float acc = 0.f;
        for (int key = 0; key < EPG; ++key)
            acc += sc[hh][key] * (float)Vp[(size_t)(b * EPG + key) * 128 + hh * DH + d];
        pooled[b * 128 + tid] = acc;
    }
}

// ---------------------------------------------------------------------------
// Kernel 6: per-graph tail: out-proj + res + LN, FF + res + LN, output MLP
// ---------------------------------------------------------------------------
__global__ __launch_bounds__(128)
void k_finalize(const float* __restrict__ seed, const float* __restrict__ pooled,
                const float* __restrict__ wo, const float* __restrict__ bo,
                const float* __restrict__ g1, const float* __restrict__ be1,
                const float* __restrict__ fw1, const float* __restrict__ fb1,
                const float* __restrict__ fw2, const float* __restrict__ fb2,
                const float* __restrict__ g2, const float* __restrict__ be2,
                const float* __restrict__ ow1, const float* __restrict__ ob1,
                const float* __restrict__ ow2, const float* __restrict__ ob2,
                float* __restrict__ out) {
    __shared__ float a0[128], a1[128], a2[128], red[128];
    int b = blockIdx.x, c = threadIdx.x;
    const float* po = pooled + b * 128;

    float t = bo[c];
    for (int k = 0; k < 128; ++k) t += po[k] * wo[k * 128 + c];
    a0[c] = seed[c] + t;
    __syncthreads();

    float mean = 0.f;
    for (int k = 0; k < 128; ++k) mean += a0[k];
    mean *= (1.f / 128.f);
    float var = 0.f;
    for (int k = 0; k < 128; ++k) { float d = a0[k] - mean; var += d * d; }
    var *= (1.f / 128.f);
    float h1 = (a0[c] - mean) * rsqrtf(var + 1e-5f) * g1[c] + be1[c];
    a1[c] = h1;
    __syncthreads();

    float r1 = fb1[c];
    for (int k = 0; k < 128; ++k) r1 += a1[k] * fw1[k * 128 + c];
    a2[c] = fmaxf(r1, 0.f);
    __syncthreads();

    float t2 = fb2[c];
    for (int k = 0; k < 128; ++k) t2 += a2[k] * fw2[k * 128 + c];
    t2 += a1[c];
    a0[c] = t2;
    __syncthreads();

    mean = 0.f;
    for (int k = 0; k < 128; ++k) mean += a0[k];
    mean *= (1.f / 128.f);
    var = 0.f;
    for (int k = 0; k < 128; ++k) { float d = a0[k] - mean; var += d * d; }
    var *= (1.f / 128.f);
    float h2 = (a0[c] - mean) * rsqrtf(var + 1e-5f) * g2[c] + be2[c];
    a1[c] = h2;
    __syncthreads();

    float m1v = ob1[c];
    for (int k = 0; k < 128; ++k) m1v += a1[k] * ow1[k * 128 + c];
    m1v = fmaxf(m1v, 0.f);
    red[c] = m1v * ow2[c];
    __syncthreads();
    if (c == 0) {
        float s = ob2[0];
        for (int k = 0; k < 128; ++k) s += red[k];
        out[b] = s;
    }
}

// ---------------------------------------------------------------------------
// Host driver
// ---------------------------------------------------------------------------
extern "C" void kernel_launch(void* const* d_in, const int* in_sizes, int n_in,
                              void* d_out, int out_size, void* d_ws, size_t ws_size,
                              hipStream_t stream) {
    (void)in_sizes; (void)n_in; (void)out_size; (void)ws_size;
    auto F = [&](int i) { return (const float*)d_in[i]; };

    const float* x  = F(0);
    const float* ea = F(1);
    const float *iw1 = F(2), *ib1 = F(3), *iw2 = F(4), *ib2 = F(5);

    struct Mab {
        const float *wq, *bq, *wk, *bk, *wv, *bv, *wo, *bo;
        const float *g1, *be1, *f1w, *f1b, *f2w, *f2b, *g2, *be2;
    };
    auto getMab = [&](int o) {
        Mab m{F(o), F(o + 1), F(o + 2), F(o + 3), F(o + 4), F(o + 5), F(o + 6), F(o + 7),
              F(o + 8), F(o + 9), F(o + 10), F(o + 11), F(o + 12), F(o + 13), F(o + 14), F(o + 15)};
        return m;
    };
    Mab m1 = getMab(6), m2 = getMab(22), ms = getMab(38), mp = getMab(54);
    const float* seed = F(70);
    const float *ow1 = F(71), *ob1 = F(72), *ow2 = F(73), *ob2 = F(74);
    const int* ei = (const int*)d_in[75];
    const int* src = ei;
    const int* dst = ei + STOT;

    float* out = (float*)d_out;
    __bf16* bws = (__bf16*)d_ws;
    const size_t SH = (size_t)STOT * 128;
    __bf16* HA   = bws + 0 * SH;
    __bf16* HB   = bws + 1 * SH;
    __bf16* QB   = bws + 2 * SH;
    __bf16* KB   = bws + 3 * SH;
    __bf16* VB   = bws + 4 * SH;
    __bf16* OBuf = bws + 5 * SH;
    __bf16* T1   = bws + 6 * SH;
    __bf16* warena = bws + 7 * SH;      // bf16 transposed-weight arena (<1 MB)
    float*  pooled = (float*)(bws + 8 * SH);

    size_t woff = 0;
    auto alloc_wt = [&](int Kp) { __bf16* p = warena + woff; woff += (size_t)128 * Kp; return p; };
    auto cvt = [&](const float* wsrc, __bf16* wdst, int K, int Kp) {
        k_wcvt<<<(128 * Kp + 255) / 256, 256, 0, stream>>>(wsrc, wdst, K, Kp);
    };

    __bf16* w1t = alloc_wt(160);  cvt(iw1, w1t, 144, 160);
    __bf16* w2t = alloc_wt(128);  cvt(iw2, w2t, 128, 128);

    struct MabW { __bf16 *wqt, *wkt, *wvt, *wot, *f1t, *f2t; };
    auto cvtMab = [&](const Mab& P) {
        MabW W;
        W.wqt = alloc_wt(128); cvt(P.wq,  W.wqt, 128, 128);
        W.wkt = alloc_wt(128); cvt(P.wk,  W.wkt, 128, 128);
        W.wvt = alloc_wt(128); cvt(P.wv,  W.wvt, 128, 128);
        W.wot = alloc_wt(128); cvt(P.wo,  W.wot, 128, 128);
        W.f1t = alloc_wt(128); cvt(P.f1w, W.f1t, 128, 128);
        W.f2t = alloc_wt(128); cvt(P.f2w, W.f2t, 128, 128);
        return W;
    };
    MabW w_m1 = cvtMab(m1), w_m2 = cvtMab(m2), w_ms = cvtMab(ms);
    __bf16* mp_wkt = alloc_wt(128); cvt(mp.wk, mp_wkt, 128, 128);
    __bf16* mp_wvt = alloc_wt(128); cvt(mp.wv, mp_wvt, 128, 128);

    k_edge_mlp<<<STOT / 64, 256, 0, stream>>>(x, ea, src, dst, w1t, ib1, w2t, ib2, HA);

    auto run_mab = [&](const __bf16* X, __bf16* Y, const Mab& P, const MabW& W, int masked) {
        k_gemm<<<STOT / 64, 256, 0, stream>>>(X, W.wqt, P.bq, nullptr, QB, 0);
        k_gemm<<<STOT / 64, 256, 0, stream>>>(X, W.wkt, P.bk, nullptr, KB, 0);
        k_gemm<<<STOT / 64, 256, 0, stream>>>(X, W.wvt, P.bv, nullptr, VB, 0);
        k_attn<<<dim3(NB, NHD, EPG / 64), 128, 0, stream>>>(QB, KB, VB, src, dst, OBuf, masked);
        k_gemm<<<STOT / 64, 256, 0, stream>>>(OBuf, W.wot, P.bo, X, T1, 0);
        k_ln<<<STOT / 8, 256, 0, stream>>>(T1, P.g1, P.be1, T1);
        k_gemm<<<STOT / 64, 256, 0, stream>>>(T1, W.f1t, P.f1b, nullptr, QB, 1);
        k_gemm<<<STOT / 64, 256, 0, stream>>>(QB, W.f2t, P.f2b, T1, Y, 0);
        k_ln<<<STOT / 8, 256, 0, stream>>>(Y, P.g2, P.be2, Y);
    };

    run_mab(HA, HB, m1, w_m1, 1);
    run_mab(HB, HA, m2, w_m2, 1);
    run_mab(HA, HB, ms, w_ms, 0);

    // PMA pooling block
    k_gemm<<<STOT / 64, 256, 0, stream>>>(HB, mp_wkt, mp.bk, nullptr, KB, 0);
    k_gemm<<<STOT / 64, 256, 0, stream>>>(HB, mp_wvt, mp.bv, nullptr, VB, 0);
    k_pma<<<NB, 256, 0, stream>>>(seed, mp.wq, mp.bq, KB, VB, pooled);
    k_finalize<<<NB, 128, 0, stream>>>(seed, pooled, mp.wo, mp.bo, mp.g1, mp.be1,
                                       mp.f1w, mp.f1b, mp.f2w, mp.f2b, mp.g2, mp.be2,
                                       ow1, ob1, ow2, ob2, out);
}
